// NormalizedCrossCorrelation2d_87531433492542
// MI455X (gfx1250) — matrix operand order repaired and checked by reading, not compiled, not run on hardware
//
#include <hip/hip_runtime.h>

typedef _Float16 v16h __attribute__((ext_vector_type(16)));
typedef float    v8f  __attribute__((ext_vector_type(8)));

#define IMG       256
#define OUTDIM    248          // 256 - 9 + 1
#define PATCH     9
#define BAND_ROWS 16           // output rows per workgroup (WMMA M)
#define IN_ROWS   24           // BAND_ROWS + PATCH - 1
#define K_PAD     32           // WMMA K (rows 24..31 zero-padded in LDS)
#define NBANDS    16           // ceil(248/16)
#define NBATCH    8
#define EPS_F     1e-4f

// One workgroup = one (batch, 16-row output band).
// Phase A (per channel): horizontal 9-tap box sums -> LDS (exact f32, VALU).
// Phase B (per channel): vertical 9-tap box sums as banded-ones matmul on
//   v_wmma_f32_16x16x32_f16; data split into exact hi/lo f16 halves so the
//   f32 accumulator carries full precision (ones side is exact in f16).
// The h-buffer is padded to K=32 rows of zeros so the B-operand gather is
// branch-free and uniform (no per-lane EXEC masking around ds_loads).
__global__ __launch_bounds__(256) void ncc_band_kernel(
    const float* __restrict__ x1, const float* __restrict__ x2,
    float* __restrict__ partials) {
  __shared__ float hbuf[K_PAD][IMG];   // 32 KB: one channel's horizontal sums

  const int tid       = threadIdx.x;
  const int b         = blockIdx.x >> 4;
  const int band      = blockIdx.x & 15;
  const int i0        = band * BAND_ROWS;
  const int lane      = tid & 31;
  const int wave      = tid >> 5;        // 8 waves; wave w owns col-tiles w, w+8
  const int laneGroup = lane >> 4;       // 0: lanes 0-15, 1: lanes 16-31
  const int n         = lane & 15;

  const float* img1 = x1 + (size_t)b * IMG * IMG;
  const float* img2 = x2 + (size_t)b * IMG * IMG;

  // Zero the K-padding rows (24..31) once; Phase A never writes them.
#pragma unroll
  for (int z = 0; z < (K_PAD - IN_ROWS); ++z) {
    hbuf[IN_ROWS + z][tid] = 0.0f;
  }

  // Banded-ones A matrix, 16x32 f16: A[m][k] = 1 iff m <= k <= m+8 (k<24 auto).
  // 16-bit A layout: lanes 0-15 M=lane, VGPR v half h -> K = 2v+h (v<4) else
  // 16+2(v-4)+h; lanes 16-31 same M, K offset +8.
  v16h Aband;
#pragma unroll
  for (int i = 0; i < 16; ++i) {
    int vv = i >> 1, hh = i & 1;
    int k = (vv < 4) ? (2 * vv + hh + 8 * laneGroup)
                     : (16 + 2 * (vv - 4) + hh + 8 * laneGroup);
    Aband[i] = (_Float16)((k >= n && k <= n + 8) ? 1.0f : 0.0f);
  }

  v8f acc[2][5] = {};   // [col-tile][channel] f32 accumulators (C/D layout)

#pragma unroll
  for (int c = 0; c < 5; ++c) {
    // ---- Phase A: horizontal 9-tap sums of channel c (rows i0..i0+23) ----
#pragma unroll 1
    for (int it = 0; it < (IN_ROWS * IMG) / 256; ++it) {
      int p = it * 256 + tid;
      int r = p >> 8, j = p & 255;
      int row = i0 + r;
      float s = 0.0f;
      if (row < IMG && j < OUTDIM) {
        const float* p1 = img1 + row * IMG + j;
        const float* p2 = img2 + row * IMG + j;
#pragma unroll
        for (int d = 0; d < PATCH; ++d) {
          float a  = p1[d];
          float bb = p2[d];
          float val = (c == 0) ? a
                    : (c == 1) ? bb
                    : (c == 2) ? a * a
                    : (c == 3) ? bb * bb
                               : a * bb;
          s += val;
        }
      }
      hbuf[r][j] = s;   // zeros outside valid region
    }
    __syncthreads();

    // ---- Phase B: vertical 9-tap sums via WMMA (banded ones x data) ----
    // 16-bit B layout (32x16): half-index i, K = i + 8*laneGroup (i<8) else
    // 16 + 8*laneGroup + (i-8); N = lane%16. Rows >= 24 are LDS zeros, so
    // the gather is unconditional and uniform.
#pragma unroll
    for (int t = 0; t < 2; ++t) {
      int col = (wave + 8 * t) * 16 + n;
      v16h Bhi, Blo;
#pragma unroll
      for (int i = 0; i < 16; ++i) {
        int k = (i < 8) ? (i + 8 * laneGroup) : (16 + 8 * laneGroup + (i - 8));
        float hv = hbuf[k][col];
        _Float16 hi = (_Float16)hv;
        _Float16 lo = (_Float16)(hv - (float)hi);
        Bhi[i] = hi;
        Blo[i] = lo;
      }
      acc[t][c] = __builtin_amdgcn_wmma_f32_16x16x32_f16(
          false, Aband, false, Bhi, (short)0, acc[t][c], false, false);
      acc[t][c] = __builtin_amdgcn_wmma_f32_16x16x32_f16(
          false, Aband, false, Blo, (short)0, acc[t][c], false, false);
    }
    __syncthreads();
  }

  // ---- Combine the five windowed sums per output position ----
  const float inv81 = 1.0f / 81.0f;
  float total = 0.0f;
#pragma unroll
  for (int t = 0; t < 2; ++t) {
    int col = (wave + 8 * t) * 16 + n;
#pragma unroll
    for (int r = 0; r < 8; ++r) {
      int orow = i0 + r + 8 * laneGroup;   // C/D layout: M = r + 8*laneGroup
      if (orow < OUTDIM && col < OUTDIM) {
        float S1  = acc[t][0][r];
        float S2  = acc[t][1][r];
        float S11 = acc[t][2][r];
        float S22 = acc[t][3][r];
        float S12 = acc[t][4][r];
        float mu1  = S1 * inv81;
        float mu2  = S2 * inv81;
        float var1 = S11 * inv81 - mu1 * mu1 + EPS_F;
        float var2 = S22 * inv81 - mu2 * mu2 + EPS_F;
        float num  = S12 - S1 * mu2;       // S12 - S1*S2/81
        total += num / sqrtf(var1 * var2);
      }
    }
  }

  // ---- Deterministic block reduction (reuse LDS) ----
  __syncthreads();
  float* red = &hbuf[0][0];
  red[tid] = total;
  __syncthreads();
#pragma unroll
  for (int s = 128; s > 0; s >>= 1) {
    if (tid < s) red[tid] += red[tid + s];
    __syncthreads();
  }
  if (tid == 0) {
    const float scale = 1.0f / (81.0f * 248.0f * 248.0f);
    partials[blockIdx.x] = red[0] * scale;
  }
}

// 128 band partials (fixed order) -> 8 batch scores. Deterministic.
__global__ void ncc_reduce_kernel(const float* __restrict__ partials,
                                  float* __restrict__ out) {
  int t = threadIdx.x;
  if (t < NBATCH) {
    float s = 0.0f;
#pragma unroll
    for (int i = 0; i < NBANDS; ++i) s += partials[t * NBANDS + i];
    out[t] = s;
  }
}

extern "C" void kernel_launch(void* const* d_in, const int* in_sizes, int n_in,
                              void* d_out, int out_size, void* d_ws, size_t ws_size,
                              hipStream_t stream) {
  const float* x1 = (const float*)d_in[0];
  const float* x2 = (const float*)d_in[1];
  float* out      = (float*)d_out;
  float* partials = (float*)d_ws;   // 128 floats of scratch

  ncc_band_kernel<<<dim3(NBATCH * NBANDS), dim3(256), 0, stream>>>(x1, x2, partials);
  ncc_reduce_kernel<<<dim3(1), dim3(32), 0, stream>>>(partials, out);

  (void)in_sizes; (void)n_in; (void)out_size; (void)ws_size;
}
